// PrototypeMatchingModel_70480413327386
// MI455X (gfx1250) — matrix operand order repaired and checked by reading, not compiled, not run on hardware
//
#include <hip/hip_runtime.h>

// ---------------------------------------------------------------------------
// PrototypeMatching for MI455X (gfx1250, wave32, WMMA)
//   x: (64, 256, 32, 32) f32   proto: (512, 256) f32
//   out: recon (64,256,32,32) f32  ++  indices (64,1024) i32
// argmax_p over cosine sims == argmax_p of x . (proto_p/||proto_p||)
// (x-normalization is a positive scalar per pixel -> argmax-invariant)
// ---------------------------------------------------------------------------

#define B_   64
#define C_   256
#define HW_  1024
#define P_   512
#define MTOT (B_ * HW_)     // 65536 pixels

typedef __attribute__((ext_vector_type(16))) __bf16 v16bf;
typedef __attribute__((ext_vector_type(8)))  float  v8f;

union Frag { v16bf v; unsigned short s[16]; uint4 u[2]; };

__device__ __forceinline__ unsigned short f2bf(float f) {
    union { float f; unsigned int u; } v; v.f = f;
    unsigned int r = v.u + 0x7FFFu + ((v.u >> 16) & 1u);   // RNE
    return (unsigned short)(r >> 16);
}

// ---------------------------------------------------------------------------
// Kernel 1: normalize prototypes, emit bf16 row-major pn[512][256] into ws.
// One wave per prototype row.
// ---------------------------------------------------------------------------
__global__ void proto_norm_kernel(const float* __restrict__ proto,
                                  unsigned short* __restrict__ pn) {
    int p    = blockIdx.x;          // 512 blocks of 32 threads
    int lane = threadIdx.x;
    const float* row = proto + p * C_;
    float vals[8];
    float ss = 0.f;
#pragma unroll
    for (int i = 0; i < 8; ++i) {
        vals[i] = row[lane + i * 32];
        ss += vals[i] * vals[i];
    }
#pragma unroll
    for (int off = 16; off > 0; off >>= 1) ss += __shfl_xor(ss, off, 32);
    float scale = 1.0f / fmaxf(sqrtf(ss), 1e-12f);
#pragma unroll
    for (int i = 0; i < 8; ++i)
        pn[p * C_ + lane + i * 32] = f2bf(vals[i] * scale);
}

// ---------------------------------------------------------------------------
// Kernel 2: fused bf16 WMMA GEMM + argmax.
// Block = 256 threads (8 waves), owns 128 pixels of one batch image.
// LDS: full bf16 prototype bank (256 KB) + fp32 half-K x tile 128x128 (64 KB)
//      = 320 KB (CDNA5 per-workgroup max, 1 WG per WGP).
// x is staged COALESCED in its native k-major layout (conflict-free LDS
// writes); the k<->pixel transpose happens once per wave while building the
// 8 A fragments (strided ds_load_b32 + bf16 cvt), which is amortized over
// the 256 WMMAs of the N sweep.
// ---------------------------------------------------------------------------
__launch_bounds__(256, 1)
__global__ void sim_argmax_kernel(const float* __restrict__ x,
                                  const unsigned short* __restrict__ pn,
                                  int* __restrict__ idxOut) {
    extern __shared__ char smem[];
    unsigned short* lds_pn = (unsigned short*)smem;             // [512][256] bf16
    float*          lds_x  = (float*)(smem + P_ * C_ * 2);      // [128 k][128 n] f32

    const int tid = threadIdx.x;
    const int b   = blockIdx.x >> 3;
    const int n0  = (blockIdx.x & 7) * 128;

    const int lane = tid & 31;
    const int w    = tid >> 5;      // wave id: 16-pixel strip
    const int lo16 = lane & 15;
    const int half = lane >> 4;
    const int mloc = w * 16 + lo16; // this lane's A row (local pixel)

    // ---- stage normalized prototype bank into LDS (coalesced b128) ----
    {
        const uint4* src = (const uint4*)pn;
        uint4*       dst = (uint4*)lds_pn;
        for (int i = tid; i < (P_ * C_ * 2) / 16; i += 256) dst[i] = src[i];
    }

    // ---- stage x in two half-K passes, build A fragments in VGPRs ----
    // A 16x32 bf16 layout: lanes 0-15: V0-3 = K0..7, V4-7 = K16..23
    //                      lanes 16-31: V0-3 = K8..15, V4-7 = K24..31
    Frag afrag[8];
    const float* xb = x + (size_t)b * C_ * HW_ + n0;
#pragma unroll
    for (int h = 0; h < 2; ++h) {
        // coalesced copy: 128 k-rows x 128 n, float4 per thread-iteration.
        // consecutive lanes -> consecutive 16B globally AND in LDS.
        {
            float4* dstv = (float4*)lds_x;
            for (int i = tid; i < 128 * 32; i += 256) {
                int k  = i >> 5;
                int c4 = i & 31;
                dstv[k * 32 + c4] =
                    *(const float4*)(xb + (size_t)(h * 128 + k) * HW_ + c4 * 4);
            }
        }
        __syncthreads();
#pragma unroll
        for (int kt = 0; kt < 4; ++kt) {
            Frag& af = afrag[h * 4 + kt];
            const int kb = kt * 32 + half * 8;
#pragma unroll
            for (int e = 0; e < 8; ++e) {
                af.s[e]     = f2bf(lds_x[(kb + e)      * 128 + mloc]);
                af.s[e + 8] = f2bf(lds_x[(kb + 16 + e) * 128 + mloc]);
            }
        }
        __syncthreads();
    }

    float best[8];
    int   bidx[8];
#pragma unroll
    for (int r = 0; r < 8; ++r) { best[r] = -__builtin_inff(); bidx[r] = 0; }

    // ---- sweep all 512 prototypes: 32 N-tiles x 8 K-steps of WMMA ----
    for (int nt = 0; nt < 32; ++nt) {
        const int   nloc = nt * 16 + lo16;  // this lane's B column (prototype)
        const char* bRow = (const char*)lds_pn + nloc * (C_ * 2);
        v8f acc = {0.f, 0.f, 0.f, 0.f, 0.f, 0.f, 0.f, 0.f};
#pragma unroll
        for (int kt = 0; kt < 8; ++kt) {
            // B 32x16 bf16 layout: lanes 0-15 hold K0..7, K16..23 of col n,
            //                      lanes 16-31 hold K8..15, K24..31
            Frag bf;
            int k0 = kt * 32 + half * 8;
            bf.u[0] = *(const uint4*)(bRow + k0 * 2);
            bf.u[1] = *(const uint4*)(bRow + k0 * 2 + 32);
            acc = __builtin_amdgcn_wmma_f32_16x16x32_bf16(
                false, afrag[kt].v, false, bf.v, (short)0, acc, false, false);
        }
        // running argmax; strict '>' keeps first (lowest) index on ties
#pragma unroll
        for (int r = 0; r < 8; ++r) {
            float v = acc[r];
            if (v > best[r]) { best[r] = v; bidx[r] = nloc; }
        }
    }

    // ---- cross-lane argmax over the 16 columns in each half-wave ----
#pragma unroll
    for (int r = 0; r < 8; ++r) {
        float bv = best[r];
        int   bi = bidx[r];
#pragma unroll
        for (int off = 1; off < 16; off <<= 1) {
            float ov = __shfl_xor(bv, off, 32);
            int   oi = __shfl_xor(bi, off, 32);
            if (ov > bv || (ov == bv && oi < bi)) { bv = ov; bi = oi; }
        }
        bidx[r] = bi;
    }

    // C layout: half 0 -> rows M=0..7, half 1 -> rows M=8..15
    if (lo16 == 0) {
        int gbase = b * HW_ + n0 + w * 16 + half * 8;
#pragma unroll
        for (int r = 0; r < 8; ++r) idxOut[gbase + r] = bidx[r];
    }
}

// ---------------------------------------------------------------------------
// Kernel 3: gather unnormalized prototypes -> recon (B, C, H, W).
// 4 pixels per thread: int4 index load + 4 L2-resident gathers + float4 store.
// ---------------------------------------------------------------------------
__global__ void gather_kernel(const float* __restrict__ proto,
                              const int* __restrict__ idx,
                              float* __restrict__ recon) {
    int t = blockIdx.x * blockDim.x + threadIdx.x;   // 4,194,304 threads
    int n4 = t & 255;                // group of 4 pixels within image row-major HW
    int c  = (t >> 8) & 255;
    int b  = t >> 16;
    int4 p4 = *(const int4*)(idx + b * HW_ + n4 * 4);
    float4 o;
    o.x = proto[p4.x * C_ + c];
    o.y = proto[p4.y * C_ + c];
    o.z = proto[p4.z * C_ + c];
    o.w = proto[p4.w * C_ + c];
    *(float4*)(recon + (size_t)b * C_ * HW_ + (size_t)c * HW_ + n4 * 4) = o;
}

extern "C" void kernel_launch(void* const* d_in, const int* in_sizes, int n_in,
                              void* d_out, int out_size, void* d_ws, size_t ws_size,
                              hipStream_t stream) {
    const float* x     = (const float*)d_in[0];   // (64,256,32,32) f32
    const float* proto = (const float*)d_in[1];   // (512,256) f32

    unsigned short* pn = (unsigned short*)d_ws;   // 512*256 bf16 = 256 KB scratch

    float* recon  = (float*)d_out;                          // 16,777,216 f32
    int*   idxOut = (int*)d_out + (size_t)B_ * C_ * HW_;    // 65,536 i32

    // 1) prototype L2-normalize -> bf16
    proto_norm_kernel<<<P_, 32, 0, stream>>>(proto, pn);

    // 2) fused WMMA similarity + argmax (320 KB dynamic LDS per workgroup)
    size_t ldsBytes = (size_t)P_ * C_ * 2 + (size_t)128 * 128 * 4; // 327,680 B
    sim_argmax_kernel<<<MTOT / 128, 256, ldsBytes, stream>>>(x, pn, idxOut);

    // 3) gather prototypes into (B, C, H, W), float4 per thread
    gather_kernel<<<(B_ * C_ * HW_) / (256 * 4), 256, 0, stream>>>(proto, idxOut, recon);
}
